// LSTMAutoencoder_28458453303916
// MI455X (gfx1250) — compile-verified, matrix operand order checked by reading, not run on hardware
//
#include <hip/hip_runtime.h>

#define BB 64
#define TT 512
#define DD 32
#define HH 256
#define GG 1024
#define RT (BB * TT)   // 32768 rows of (b*T + t)
#define NWG 16

typedef __attribute__((ext_vector_type(8)))  __bf16 v8bf;
typedef __attribute__((ext_vector_type(16))) __bf16 v16bf;
typedef __attribute__((ext_vector_type(8)))  float  v8f;
typedef __attribute__((ext_vector_type(4)))  int    v4i;

#define AS1 __attribute__((address_space(1)))
#define AS3 __attribute__((address_space(3)))

#if __has_builtin(__builtin_amdgcn_global_load_async_to_lds_b128)
#define HAVE_ASYNC_LDS 1
#else
#define HAVE_ASYNC_LDS 0
#endif

#if HAVE_ASYNC_LDS
// per-lane 16B global -> LDS async copy (ASYNCcnt-tracked, no VGPR round trip)
static __device__ __forceinline__ void async_cp16(const void* g, void* l) {
  __builtin_amdgcn_global_load_async_to_lds_b128(
      (AS1 v4i*)(uintptr_t)g,          // AS1 == generic global address
      (AS3 v4i*)(v4i*)l,               // proper addrspacecast for LDS
      0, 0);
}
#endif

static __device__ __forceinline__ void wait_async0() {
#if __has_builtin(__builtin_amdgcn_s_wait_asynccnt)
  __builtin_amdgcn_s_wait_asynccnt(0);
#else
  asm volatile("s_wait_asynccnt 0x0" ::: "memory");
#endif
}

static __device__ __forceinline__ __bf16 f2bf(float f) {
  unsigned u = __builtin_bit_cast(unsigned, f);
  u += 0x7FFFu + ((u >> 16) & 1u);               // round-to-nearest-even
  unsigned short s = (unsigned short)(u >> 16);
  return __builtin_bit_cast(__bf16, s);
}
static __device__ __forceinline__ float bf2f(__bf16 b) {
  unsigned short s = __builtin_bit_cast(unsigned short, b);
  unsigned u = ((unsigned)s) << 16;
  return __builtin_bit_cast(float, u);
}
static __device__ __forceinline__ float sigmoidf_(float x) {
  return 1.0f / (1.0f + __expf(-x));
}
static __device__ __forceinline__ float tanhf_(float x) {
  return 2.0f / (1.0f + __expf(-2.0f * x)) - 1.0f;   // v_exp_f32 based
}

// ---------------- fp32 -> bf16 conversion ----------------
__global__ void cvt_bf16_kernel(const float* __restrict__ in,
                                __bf16* __restrict__ out, int n) {
  int i = blockIdx.x * 256 + threadIdx.x;
  if (i < n) out[i] = f2bf(in[i]);
}

// ---------------- tiled GEMM: out[R,N] = A[R,K](bf16) x W[N,K](bf16)^T + bias ----------------
// Block 256 thr = 8 waves; tile 64(M) x 128(N); K sliced by 32 with LDS double buffering
// via CDNA5 async global->LDS copies (ASYNCcnt) when available.
// LDS rows padded to 40 el (80B: bank stride 20 dwords -> conflict-free fragment reads).
__global__ __launch_bounds__(256) void gemm_wmma_kernel(
    const __bf16* __restrict__ A, const __bf16* __restrict__ W,
    const float* __restrict__ bias, float* __restrict__ out,
    int K, int N, int ldo) {
  __shared__ __bf16 sA[2][64 * 40];    // 10.0 KB
  __shared__ __bf16 sB[2][128 * 40];   // 20.0 KB
  const int tid  = threadIdx.x;
  const int lane = tid & 31;
  const int wave = tid >> 5;
  const int mt   = wave & 3;
  const int ngrp = wave >> 2;
  const int row0 = blockIdx.x * 64;
  const int ncol0 = blockIdx.y * 128;
  const int nl   = lane & 15;
  const int koff = (lane & 16) ? 8 : 0;

  const int ar = tid >> 2;             // A staging: 64 rows x 4 chunks of 8 el
  const int ac = (tid & 3) * 8;

#if HAVE_ASYNC_LDS
  auto stage_async = [&](int k0, int buf) {
    async_cp16(A + (size_t)(row0 + ar) * K + k0 + ac, &sA[buf][ar * 40 + ac]);
#pragma unroll
    for (int s = 0; s < 2; ++s) {      // B staging: 128 rows x 4 chunks, 2 per thread
      int idx = s * 256 + tid;
      int r = idx >> 2, c = (idx & 3) * 8;
      int nn = ncol0 + r;
      if (nn >= N) nn = N - 1;         // guard (head: N=32)
      async_cp16(W + (size_t)nn * K + k0 + c, &sB[buf][r * 40 + c]);
    }
  };
#else
  v8bf ra;
  v8bf rb[2];
  auto gload = [&](int k0) {
    ra = *(const v8bf*)(A + (size_t)(row0 + ar) * K + k0 + ac);
#pragma unroll
    for (int s = 0; s < 2; ++s) {
      int idx = s * 256 + tid;
      int r = idx >> 2, c = (idx & 3) * 8;
      int nn = ncol0 + r;
      if (nn >= N) nn = N - 1;
      rb[s] = *(const v8bf*)(W + (size_t)nn * K + k0 + c);
    }
  };
  auto lstore = [&](int buf) {
    *(v8bf*)&sA[buf][ar * 40 + ac] = ra;
#pragma unroll
    for (int s = 0; s < 2; ++s) {
      int idx = s * 256 + tid;
      int r = idx >> 2, c = (idx & 3) * 8;
      *(v8bf*)&sB[buf][r * 40 + c] = rb[s];
    }
  };
#endif

#if HAVE_ASYNC_LDS
  stage_async(0, 0);
  wait_async0();
  __syncthreads();
#else
  gload(0);
  lstore(0);
  __syncthreads();
#endif

  v8f acc[4] = {};
  const int nk = K >> 5;
  for (int ks = 0; ks < nk; ++ks) {
    const int cur = ks & 1;
#if HAVE_ASYNC_LDS
    if (ks + 1 < nk) stage_async((ks + 1) * 32, cur ^ 1);  // async fill of other buffer
#else
    if (ks + 1 < nk) gload((ks + 1) * 32);                 // overlap fetch with compute
#endif

    v16bf a;
    {
      const __bf16* pa = &sA[cur][(mt * 16 + nl) * 40 + koff];
      ((v8bf*)&a)[0] = *(const v8bf*)(pa);
      ((v8bf*)&a)[1] = *(const v8bf*)(pa + 16);
    }
    v16bf bfr[4];
#pragma unroll
    for (int j = 0; j < 4; ++j) {
      const __bf16* pb = &sB[cur][(ngrp * 64 + j * 16 + nl) * 40 + koff];
      ((v8bf*)&bfr[j])[0] = *(const v8bf*)(pb);
      ((v8bf*)&bfr[j])[1] = *(const v8bf*)(pb + 16);
    }
#pragma unroll
    for (int j = 0; j < 4; ++j) {
      acc[j] = __builtin_amdgcn_wmma_f32_16x16x32_bf16(
          false, a, false, bfr[j], (short)0, acc[j], false, false);
    }

#if HAVE_ASYNC_LDS
    if (ks + 1 < nk) wait_async0();    // own async copies into other buffer done
#else
    if (ks + 1 < nk) lstore(cur ^ 1);
#endif
    __syncthreads();
  }

#pragma unroll
  for (int j = 0; j < 4; ++j) {
    int nn = ncol0 + ngrp * 64 + j * 16 + nl;
    if (nn < N) {
      float bs = bias[nn];
#pragma unroll
      for (int i = 0; i < 8; ++i) {
        int r = row0 + mt * 16 + ((lane & 16) ? 8 + i : i);
        out[(size_t)r * ldo + nn] = acc[j][i] + bs;
      }
    }
  }
}

// ---------------- encoder output: compact relu(last timestep) -> rep[BB,HH] ----------------
__global__ void relu_last_kernel(const __bf16* __restrict__ yin,
                                 __bf16* __restrict__ rep) {
  int i = blockIdx.x * 256 + threadIdx.x;    // over BB*HH = 16384, exact grid
  int b = i >> 8;
  int hc = i & (HH - 1);
  float v = bf2f(yin[((size_t)b * TT + (TT - 1)) * HH + hc]);
  rep[i] = f2bf(v > 0.f ? v : 0.f);
}

// ---------------- recurrent scan: 16 persistent WGs, each owns 16 h-columns ----------------
// 128 threads = 4 waves; wave = M tile; each wave owns ALL 4 gate tiles of its rows, so
// gates, cell state and the LSTM elementwise live entirely in registers.
// xp row index = b*rmb + t*rmt (rmb=TT,rmt=1 normal; rmb=1,rmt=0 for the broadcast decoder
// input, whose projection is computed once for 64 rows instead of 32768).
__global__ __launch_bounds__(128) void lstm_scan_kernel(
    const float* __restrict__ xp,    // preactivations (x·Wih^T + b)
    const __bf16* __restrict__ Whh,  // [GG, HH] bf16
    __bf16* __restrict__ y,          // [RT, HH] output h sequence
    __bf16* __restrict__ hg,         // [2, BB, HH] double-buffered h (zeroed)
    unsigned int* __restrict__ bar,  // monotonic counter (zeroed)
    int rmb, int rmt)
{
  __shared__ __bf16 sW[64 * 264];  // Whh slice: rows = gate*16+hcol, K contig, padded stride

  const int wg   = blockIdx.x;
  const int tid  = threadIdx.x;
  const int lane = tid & 31;
  const int mt   = tid >> 5;       // wave index = M tile
  const int nl   = lane & 15;
  const int koff = (lane & 16) ? 8 : 0;
  const int colg = wg * 16 + nl;   // this lane's h column

  // Stage this WG's Whh slice (4 gates x 16 h-cols x 256 K) into LDS once.
  for (int idx = tid; idx < 64 * 32; idx += 128) {
    int ln = idx >> 5;             // local row 0..63
    int ck = idx & 31;             // 16B chunk
    int q  = ln >> 4;              // gate 0..3
    int nc = ln & 15;
    *(v8bf*)&sW[ln * 264 + ck * 8] =
        *(const v8bf*)(Whh + (size_t)(q * HH + wg * 16 + nc) * HH + ck * 8);
  }
  __syncthreads();

  v8f creg = {};                   // cell state, 8 elements per lane
  for (int t = 0; t < TT; ++t) {
    const __bf16* hr = hg + (size_t)(t & 1) * (BB * HH);        // read buffer
    __bf16* hw       = hg + (size_t)((t + 1) & 1) * (BB * HH);  // write buffer

    // Preload ALL A fragments for this step: 16 global_load_b128 in flight at once,
    // so the WMMA chain below only pays one L2 round-trip, not eight.
    v16bf afr[8];
#pragma unroll
    for (int ks = 0; ks < 8; ++ks) {
      const __bf16* pa = hr + (size_t)(mt * 16 + nl) * HH + ks * 32 + koff;
      ((v8bf*)&afr[ks])[0] = *(const v8bf*)(pa);
      ((v8bf*)&afr[ks])[1] = *(const v8bf*)(pa + 16);
    }

    // xp preactivations for this lane's 8 rows x 4 gates; loads overlap the WMMA chain
    float xg[4][8];
#pragma unroll
    for (int j = 0; j < 4; ++j)
#pragma unroll
      for (int i = 0; i < 8; ++i) {
        int b = mt * 16 + ((lane & 16) ? 8 + i : i);
        size_t row = (size_t)b * rmb + (size_t)t * rmt;
        xg[j][i] = xp[row * GG + j * HH + colg];
      }

    // gates = h_prev x Whh^T : 4 independent accumulator chains (i,f,g,o)
    v8f acc[4] = {};
#pragma unroll
    for (int ks = 0; ks < 8; ++ks) {
      int k0 = ks * 32;
      v16bf bfr[4];
#pragma unroll
      for (int j = 0; j < 4; ++j) {
        const __bf16* pb = &sW[(j * 16 + nl) * 264 + k0 + koff];
        ((v8bf*)&bfr[j])[0] = *(const v8bf*)(pb);
        ((v8bf*)&bfr[j])[1] = *(const v8bf*)(pb + 16);
      }
#pragma unroll
      for (int j = 0; j < 4; ++j) {
        acc[j] = __builtin_amdgcn_wmma_f32_16x16x32_bf16(
            false, afr[ks], false, bfr[j], (short)0, acc[j], false, false);
      }
    }

    // LSTM cell, entirely in registers (fp32 gates/cell, exp-based trans ops)
#pragma unroll
    for (int i = 0; i < 8; ++i) {
      int b = mt * 16 + ((lane & 16) ? 8 + i : i);
      float gi = acc[0][i] + xg[0][i];
      float gf = acc[1][i] + xg[1][i];
      float gc = acc[2][i] + xg[2][i];
      float go = acc[3][i] + xg[3][i];
      float c  = sigmoidf_(gf) * creg[i] + sigmoidf_(gi) * tanhf_(gc);
      creg[i] = c;
      float h  = sigmoidf_(go) * tanhf_(c);
      __bf16 hb = f2bf(h);
      hw[b * HH + colg] = hb;
      y[(size_t)(b * TT + t) * HH + colg] = hb;
    }

    // Device-scope barrier across the NWG persistent WGs (monotonic counter).
    __builtin_amdgcn_fence(__ATOMIC_RELEASE, "agent");  // release h stores (all threads)
    __syncthreads();
    if (tid == 0) {
      atomicAdd(bar, 1u);
      unsigned target = (unsigned)(t + 1) * NWG;
      while (__hip_atomic_load(bar, __ATOMIC_RELAXED, __HIP_MEMORY_SCOPE_AGENT) < target) {
        __builtin_amdgcn_s_sleep(1);
      }
    }
    __syncthreads();
    __builtin_amdgcn_fence(__ATOMIC_ACQUIRE, "agent"); // acquire other WGs' h stores
  }
}

// ---------------- host orchestration ----------------
extern "C" void kernel_launch(void* const* d_in, const int* in_sizes, int n_in,
                              void* d_out, int out_size, void* d_ws, size_t ws_size,
                              hipStream_t stream) {
  const float* x     = (const float*)d_in[0];
  const float* eWih0 = (const float*)d_in[1];
  const float* eWhh0 = (const float*)d_in[2];
  const float* eb0   = (const float*)d_in[3];
  const float* eWih1 = (const float*)d_in[4];
  const float* eWhh1 = (const float*)d_in[5];
  const float* eb1   = (const float*)d_in[6];
  const float* dWih  = (const float*)d_in[7];
  const float* dWhh  = (const float*)d_in[8];
  const float* db    = (const float*)d_in[9];
  const float* Wout  = (const float*)d_in[10];
  const float* bout  = (const float*)d_in[11];

  char* p = (char*)d_ws;
  auto carve = [&](size_t bytes) {
    char* r = p;
    p += (bytes + 255) & ~(size_t)255;
    return r;
  };
  __bf16* xbf   = (__bf16*)carve((size_t)RT * DD * 2);
  __bf16* wih0  = (__bf16*)carve((size_t)GG * DD * 2);
  __bf16* whh0  = (__bf16*)carve((size_t)GG * HH * 2);
  __bf16* wih1  = (__bf16*)carve((size_t)GG * HH * 2);
  __bf16* whh1  = (__bf16*)carve((size_t)GG * HH * 2);
  __bf16* dwih  = (__bf16*)carve((size_t)4 * GG * HH * 2);
  __bf16* dwhh  = (__bf16*)carve((size_t)4 * GG * HH * 2);
  __bf16* woutb = (__bf16*)carve((size_t)DD * HH * 2);
  __bf16* rep   = (__bf16*)carve((size_t)BB * HH * 2);       // compact relu(last step)
  __bf16* hg    = (__bf16*)carve((size_t)2 * BB * HH * 2);   // double-buffered h
  unsigned* bar = (unsigned*)carve(256);                      // contiguous after hg
  __bf16* yA    = (__bf16*)carve((size_t)RT * HH * 2);
  __bf16* yB    = (__bf16*)carve((size_t)RT * HH * 2);
  float* xp     = (float*)carve((size_t)RT * GG * 4);

  auto cvt = [&](const float* src, __bf16* dst, int n) {
    cvt_bf16_kernel<<<(n + 255) / 256, 256, 0, stream>>>(src, dst, n);
  };
  cvt(x, xbf, RT * DD);
  cvt(eWih0, wih0, GG * DD);
  cvt(eWhh0, whh0, GG * HH);
  cvt(eWih1, wih1, GG * HH);
  cvt(eWhh1, whh1, GG * HH);
  cvt(dWih, dwih, 4 * GG * HH);
  cvt(dWhh, dwhh, 4 * GG * HH);
  cvt(Wout, woutb, DD * HH);

  auto scan = [&](const __bf16* whh, __bf16* yout, int rmb, int rmt) {
    (void)hipMemsetAsync(hg, 0, (size_t)2 * BB * HH * 2 + 256, stream);  // h + barrier
    lstm_scan_kernel<<<NWG, 128, 0, stream>>>(xp, whh, yout, hg, bar, rmb, rmt);
  };
  auto layer = [&](const __bf16* Ain, int K, const __bf16* wih, const float* b,
                   const __bf16* whh, __bf16* yout) {
    gemm_wmma_kernel<<<dim3(RT / 64, GG / 128), 256, 0, stream>>>(
        Ain, wih, b, xp, K, GG, GG);
    scan(whh, yout, TT, 1);
  };

  layer(xbf, DD, wih0, eb0, whh0, yA);                                     // enc L0
  layer(yA, HH, wih1, eb1, whh1, yB);                                      // enc L1

  // decoder input: relu(last h) broadcast; projection computed ONCE (64 rows, not 32768)
  relu_last_kernel<<<BB * HH / 256, 256, 0, stream>>>(yB, rep);
  gemm_wmma_kernel<<<dim3(1, GG / 128), 256, 0, stream>>>(
      rep, dwih + 0 * GG * HH, db + 0 * GG, xp, HH, GG, GG);               // xp[64,1024]
  scan(dwhh + 0 * GG * HH, yB, 1, 0);                                      // dec1 L0

  layer(yB, HH, dwih + 1 * GG * HH, db + 1 * GG, dwhh + 1 * GG * HH, yA);  // dec1 L1
  layer(yA, HH, dwih + 2 * GG * HH, db + 2 * GG, dwhh + 2 * GG * HH, yB);  // dec2 L0
  layer(yB, HH, dwih + 3 * GG * HH, db + 3 * GG, dwhh + 3 * GG * HH, yA);  // dec2 L1
  // time-distributed head: [RT,256] x [32,256]^T + b_out -> d_out fp32 [B,T,D]
  gemm_wmma_kernel<<<dim3(RT / 64, 1), 256, 0, stream>>>(
      yA, woutb, bout, (float*)d_out, HH, DD, DD);

  (void)in_sizes; (void)n_in; (void)out_size; (void)ws_size;
}
